// DeformConvTranspose_67267777790297
// MI455X (gfx1250) — compile-verified, matrix operand order checked by reading, not run on hardware
//
#include <hip/hip_runtime.h>

#define B_   8
#define CIN  128
#define COUT 128
#define HIN  64
#define WIN  64
#define KS_  4
#define NT   16       // taps = KS*KS
#define HO   128
#define WO   128
#define OM_CH 48
#define MT_  64       // pixels (M) per block
#define NMT  4        // M subtiles of 16

typedef __attribute__((ext_vector_type(16))) _Float16 v16h;
typedef __attribute__((ext_vector_type(8)))  float    v8f;
typedef __attribute__((ext_vector_type(4)))  int      i32x4;

#if __has_builtin(__builtin_amdgcn_global_load_async_to_lds_b128) && \
    __has_builtin(__builtin_amdgcn_s_wait_asynccnt)
#define HAS_ASYNC_LDS 1
#else
#define HAS_ASYNC_LDS 0
#endif

__device__ __forceinline__ void copy_b128_to_lds(const _Float16* gsrc, _Float16* ldst) {
#if HAS_ASYNC_LDS
  __builtin_amdgcn_global_load_async_to_lds_b128(
      (__attribute__((address_space(1))) i32x4*)gsrc,
      (__attribute__((address_space(3))) i32x4*)ldst, 0, 0);
#else
  *(uint4*)ldst = *(const uint4*)gsrc;
#endif
}

__device__ __forceinline__ void async_copy_wait() {
#if HAS_ASYNC_LDS
  __builtin_amdgcn_s_wait_asynccnt(0);
#endif
}

// ---------------------------------------------------------------------------
// Kernel A: offset/mask conv over the dilated input (dilation sparsity: only
// even taps within [2,128] touch data).  One block per (b, y) row.
// ---------------------------------------------------------------------------
__global__ void om_conv_kernel(const float* __restrict__ x,
                               const float* __restrict__ w_om,
                               const float* __restrict__ b_om,
                               float* __restrict__ om) {
  __shared__ float xlds[CIN * WIN];     // one input row, all channels (32 KB)
  __shared__ float wlds[OM_CH * CIN];   // w_om[:, :, i, j]          (24 KB)
  const int tid = threadIdx.x;
  const int b = blockIdx.x / HO;
  const int y = blockIdx.x % HO;

  float acc[24];
#pragma unroll
  for (int e = 0; e < 24; ++e) {
    int oc = (tid + e * 256) >> 7;
    acc[e] = b_om[oc];
  }

  for (int i = 0; i < KS_; ++i) {
    int Y = y + i;                                   // uniform per block
    if ((Y & 1) || Y < 2 || Y > 128) continue;
    int ry = (Y - 2) >> 1;
    __syncthreads();
    for (int e = tid; e < CIN * WIN; e += 256) {
      int c = e >> 6, xx = e & 63;
      xlds[e] = x[((size_t)(b * CIN + c) * HIN + ry) * WIN + xx];
    }
    for (int j = 0; j < KS_; ++j) {
      __syncthreads();
      for (int e = tid; e < OM_CH * CIN; e += 256) {
        int oc = e >> 7, c = e & 127;
        wlds[e] = w_om[((oc * CIN + c) * KS_ + i) * KS_ + j];
      }
      __syncthreads();
#pragma unroll
      for (int e = 0; e < 24; ++e) {
        int idx = tid + e * 256;
        int oc = idx >> 7, xx = idx & 127;
        int X = xx + j;
        if ((X & 1) || X < 2 || X > 128) continue;
        int rx = (X - 2) >> 1;
        float s = 0.f;
#pragma unroll 8
        for (int c = 0; c < CIN; ++c)
          s += xlds[c * WIN + rx] * wlds[oc * CIN + c];
        acc[e] += s;
      }
    }
  }
#pragma unroll
  for (int e = 0; e < 24; ++e) {
    int idx = tid + e * 256;
    int oc = idx >> 7, xx = idx & 127;
    om[((size_t)(b * OM_CH + oc) * HO + y) * WO + xx] = acc[e];
  }
}

// ---------------------------------------------------------------------------
// Kernel B: in-place softmax over the 16 mask channels (ch 32..47).
// ---------------------------------------------------------------------------
__global__ void softmax_kernel(float* __restrict__ om) {
  int t = blockIdx.x * blockDim.x + threadIdx.x;
  if (t >= B_ * HO * WO) return;
  int b = t / (HO * WO);
  int rem = t % (HO * WO);
  float* p = om + (size_t)(b * OM_CH + 32) * HO * WO + rem;
  float v[16];
  float mx = -1e30f;
#pragma unroll
  for (int k = 0; k < 16; ++k) { v[k] = p[(size_t)k * HO * WO]; mx = fmaxf(mx, v[k]); }
  float s = 0.f;
#pragma unroll
  for (int k = 0; k < 16; ++k) { v[k] = __expf(v[k] - mx); s += v[k]; }
  float inv = 1.f / s;
#pragma unroll
  for (int k = 0; k < 16; ++k) p[(size_t)k * HO * WO] = v[k] * inv;
}

// ---------------------------------------------------------------------------
// Kernel C: flip weights spatially, transpose to [tap][o][c], convert to f16.
// ---------------------------------------------------------------------------
__global__ void wprep_kernel(const float* __restrict__ weight,
                             _Float16* __restrict__ wfT) {
  int idx = blockIdx.x * blockDim.x + threadIdx.x;
  if (idx >= NT * COUT * CIN) return;
  int k = idx >> 14;
  int o = (idx >> 7) & 127;
  int c = idx & 127;
  int i = k >> 2, j = k & 3;
  wfT[idx] = (_Float16)weight[((o * CIN + c) * KS_ + (3 - i)) * KS_ + (3 - j)];
}

// ---------------------------------------------------------------------------
// Kernel X: NCHW -> NHWC transpose of x so bilinear gathers coalesce over c.
// ---------------------------------------------------------------------------
__global__ void xprep_kernel(const float* __restrict__ x,
                             float* __restrict__ xT) {
  int idx = blockIdx.x * blockDim.x + threadIdx.x;
  if (idx >= B_ * CIN * HIN * WIN) return;
  int xx = idx & 63;
  int yy = (idx >> 6) & 63;
  int c  = (idx >> 12) & 127;
  int b  = idx >> 19;
  xT[(((size_t)b * HIN + yy) * WIN + xx) * CIN + c] = x[idx];
}

// ---------------------------------------------------------------------------
// Kernel D: deformable gather + WMMA GEMM.
// Block = 64 pixels x 128 Cout.  8 waves, each owns one 16-wide N tile across
// 4 M subtiles (4 x v8f accumulators).  B tiles are double-buffered in LDS via
// async global->LDS copies (ASYNCcnt) overlapped with the A gather + WMMAs.
// ---------------------------------------------------------------------------
__global__ void __launch_bounds__(256)
deform_wmma_kernel(const float* __restrict__ xT,
                   const _Float16* __restrict__ wfT,
                   const float* __restrict__ om,
                   const float* __restrict__ bias,
                   float* __restrict__ out) {
  __shared__ _Float16 Alds[MT_ * 136];       // 64 px x 128 Cin   (17.4 KB)
  __shared__ _Float16 Blds[2][128 * 136];    // double-buffered B (68 KB)
  __shared__ float wgt_s[MT_][4];
  __shared__ int   srcP_s[MT_][4];           // flattened (sy*WIN+sx) or -1

  const int tid  = threadIdx.x;
  const int lane = tid & 31;
  const int wave = tid >> 5;                 // N tile 0..7
  const int bx   = blockIdx.x;
  const int tile = bx & 1;
  const int y    = (bx >> 1) & 127;
  const int b    = bx >> 8;
  const int xbase = tile * MT_;

  const int mrow = lane & 15;                // A row / B column within tile
  const int hsel = lane >> 4;                // K-half select per WMMA layout

  v8f acc[NMT] = {};

  // Prologue: stage B tile for tap 0 into buffer 0.
  {
    const _Float16* src = wfT;
#pragma unroll
    for (int e = 0; e < 8; ++e) {
      int idx = tid + e * 256;               // 2048 chunks of 8 halves
      int o = idx >> 4, cb = (idx & 15) * 8;
      copy_b128_to_lds(src + o * CIN + cb, &Blds[0][o * 136 + cb]);
    }
  }
  async_copy_wait();
  __syncthreads();

  for (int k = 0; k < NT; ++k) {
    const int ki = k >> 2, kj = k & 3;
    const int cur = k & 1, nxt = cur ^ 1;

    // Issue async staging of the NEXT tap's B tile (overlaps A build + WMMA).
    if (k + 1 < NT) {
      const _Float16* src = wfT + (size_t)(k + 1) * COUT * CIN;
#if !HAS_ASYNC_LDS
      __builtin_prefetch(src + tid * 8, 0, 1);   // global_prefetch_b8 fallback
#endif
#pragma unroll
      for (int e = 0; e < 8; ++e) {
        int idx = tid + e * 256;
        int o = idx >> 4, cb = (idx & 15) * 8;
        copy_b128_to_lds(src + o * CIN + cb, &Blds[nxt][o * 136 + cb]);
      }
    }

    // Per-pixel bilinear corner precompute; mask folded into the 4 weights.
    {
      int m = tid >> 2, q = tid & 3;
      int xp = xbase + m;
      size_t pix = (size_t)y * WO + xp;
      float offY = om[(size_t)(b * OM_CH + 2 * k    ) * HO * WO + pix];
      float offX = om[(size_t)(b * OM_CH + 2 * k + 1) * HO * WO + pix];
      float msk  = om[(size_t)(b * OM_CH + 32 + k   ) * HO * WO + pix];
      float py = (float)(y  + ki) + offY;
      float px = (float)(xp + kj) + offX;
      float y0 = floorf(py), x0 = floorf(px);
      float wy1 = py - y0, wx1 = px - x0;
      int dy = q >> 1, dx = q & 1;
      int Y = (int)y0 + dy, X = (int)x0 + dx;
      float w = (dy ? wy1 : 1.f - wy1) * (dx ? wx1 : 1.f - wx1) * msk;
      int sp = -1;
      // dilated input is nonzero only at even coords in [2,128]
      if (Y >= 2 && Y <= 128 && !(Y & 1) && X >= 2 && X <= 128 && !(X & 1))
        sp = ((Y - 2) >> 1) * WIN + ((X - 2) >> 1);
      wgt_s[m][q] = w; srcP_s[m][q] = sp;
    }
    __syncthreads();

    // Build A tile: coalesced NHWC gather (consecutive lanes -> consecutive c).
    const float* xb = xT + (size_t)b * HIN * WIN * CIN;
#pragma unroll
    for (int e = 0; e < 32; ++e) {
      int idx = tid + e * 256;               // 8192 = 64 px x 128 c
      int m = idx >> 7, c = idx & 127;
      float v = 0.f;
#pragma unroll
      for (int q = 0; q < 4; ++q) {
        int sp = srcP_s[m][q];
        if (sp >= 0) v += wgt_s[m][q] * xb[(size_t)sp * CIN + c];
      }
      Alds[m * 136 + c] = (_Float16)v;
    }
    __syncthreads();

    // 4 K-chunks of 32; each B fragment reused across 4 M subtiles.
#pragma unroll
    for (int kc = 0; kc < 4; ++kc) {
      v16h bf;
      {
        const _Float16* pb = &Blds[cur][(wave * 16 + mrow) * 136 + kc * 32 + hsel * 8];
        __builtin_memcpy(&bf, pb, 16);
        __builtin_memcpy((char*)&bf + 16, pb + 16, 16);
      }
#pragma unroll
      for (int mt = 0; mt < NMT; ++mt) {
        v16h a;
        const _Float16* pa = &Alds[(mt * 16 + mrow) * 136 + kc * 32 + hsel * 8];
        __builtin_memcpy(&a, pa, 16);
        __builtin_memcpy((char*)&a + 16, pa + 16, 16);
        acc[mt] = __builtin_amdgcn_wmma_f32_16x16x32_f16(
            false, a, false, bf, (short)0, acc[mt], false, false);
      }
    }

    // Next tap's B tile must be resident before anyone reads it; A/precompute
    // buffers must be consumed before they are overwritten.
    async_copy_wait();
    __syncthreads();
  }

  // Write C tiles per the documented 16x16 f32 layout (+bias).
  const int oo = wave * 16 + mrow;
  const float bz = bias[oo];
#pragma unroll
  for (int mt = 0; mt < NMT; ++mt) {
#pragma unroll
    for (int r = 0; r < 8; ++r) {
      int m = mt * 16 + r + hsel * 8;
      out[(((size_t)b * COUT + oo) * HO + y) * WO + (xbase + m)] = acc[mt][r] + bz;
    }
  }
}

// ---------------------------------------------------------------------------
extern "C" void kernel_launch(void* const* d_in, const int* in_sizes, int n_in,
                              void* d_out, int out_size, void* d_ws, size_t ws_size,
                              hipStream_t stream) {
  const float* x      = (const float*)d_in[0];
  const float* weight = (const float*)d_in[1];
  const float* bias   = (const float*)d_in[2];
  const float* w_om   = (const float*)d_in[3];
  const float* b_om   = (const float*)d_in[4];
  float* out = (float*)d_out;

  const size_t om_bytes  = (size_t)B_ * OM_CH * HO * WO * sizeof(float);   // 25.2 MB
  const size_t wft_bytes = (size_t)NT * COUT * CIN * sizeof(_Float16);     // 512 KB
  float*    om  = (float*)d_ws;
  _Float16* wfT = (_Float16*)((char*)d_ws + om_bytes);
  float*    xT  = (float*)((char*)d_ws + om_bytes + wft_bytes);            // 16.8 MB

  om_conv_kernel<<<B_ * HO, 256, 0, stream>>>(x, w_om, b_om, om);
  softmax_kernel<<<(B_ * HO * WO + 255) / 256, 256, 0, stream>>>(om);
  wprep_kernel<<<(NT * COUT * CIN + 255) / 256, 256, 0, stream>>>(weight, wfT);
  xprep_kernel<<<(B_ * CIN * HIN * WIN + 255) / 256, 256, 0, stream>>>(x, xT);
  deform_wmma_kernel<<<B_ * HO * (WO / MT_), 256, 0, stream>>>(xT, wfT, om, bias, out);
}
